// SupervisedContrastiveLoss_23167053595533
// MI455X (gfx1250) — matrix operand order repaired and checked by reading, not compiled, not run on hardware
//
#include <hip/hip_runtime.h>
#include <hip/hip_bf16.h>
#include <math.h>

// ---- problem constants (reference: B=4096, D=128, N=2B=8192) ----
#define DDIM    128
#define BROWS   4096
#define NROWS   8192
#define NTILES  (NROWS / 16)        // 512
#define PPART   8                   // partitions: p owns jt % 16 in {2p, 2p+1}
#define NITER   32                  // 32 iterations x 2 sub-tiles = 64 jt tiles
#define LDS_STRIDE 136              // 128 + 8 halfs pad: conflict-free b128 reads
#define EXP2_SCALE 14.426950408889634f   // (1/tau) * log2(e) = 10*log2(e)
#define LN2        0.6931471805599453f
#define EPSN    1e-8f

typedef _Float16 v4h  __attribute__((ext_vector_type(4)));
typedef _Float16 v8h  __attribute__((ext_vector_type(8)));
typedef _Float16 v16h __attribute__((ext_vector_type(16)));
typedef float    v4f  __attribute__((ext_vector_type(4)));
typedef float    v8f  __attribute__((ext_vector_type(8)));

union F16x16 { v16h v; v8h h[2]; };

// workspace layout (bytes):
//   pn     : 8192*128 f16  = 2 MiB         @ 0
//   denom  : 8192 f32      = 32 KiB        @ 2 MiB   (FULL row sums incl diag)
//   S      : 2*128 f32 + cnt[2]            @ 2 MiB + 32 KiB
#define WS_PN(ws)    ((_Float16*)(ws))
#define WS_DENOM(ws) ((float*)((char*)(ws) + (2u<<20)))
#define WS_S(ws)     ((float*)((char*)(ws) + (2u<<20) + (32u<<10)))

// ---------------------------------------------------------------------------
// K1: row-normalize p = concat(z_i, z_j) -> f16 Pn; zero denom/S/cnt/out.
// One wave32 per row.
// ---------------------------------------------------------------------------
__global__ __launch_bounds__(256) void scl_normalize(
    const float* __restrict__ zi, const float* __restrict__ zj,
    _Float16* __restrict__ pn, float* __restrict__ denom,
    float* __restrict__ S, float* __restrict__ out)
{
  if (blockIdx.x == 0) {
    S[threadIdx.x] = 0.0f;                             // 256 class-sum slots
    if (threadIdx.x < 2) S[256 + threadIdx.x] = 0.0f;  // cnt[2]
    if (threadIdx.x == 0) out[0] = 0.0f;
  }
  const int wave = threadIdx.x >> 5;
  const int lane = threadIdx.x & 31;
  const int row  = blockIdx.x * 8 + wave;              // 1024 blocks * 8 = 8192

  if (lane == 0) denom[row] = 0.0f;

  const float* src = (row < BROWS) ? (zi + (size_t)row * DDIM)
                                   : (zj + (size_t)(row - BROWS) * DDIM);
  v4f v = *(const v4f*)(src + lane * 4);
  float ss = v.x * v.x + v.y * v.y + v.z * v.z + v.w * v.w;
  #pragma unroll
  for (int m = 1; m < 32; m <<= 1) ss += __shfl_xor(ss, m);
  float rn = 1.0f / fmaxf(sqrtf(ss), EPSN);

  v4h h;
  h.x = (_Float16)(v.x * rn); h.y = (_Float16)(v.y * rn);
  h.z = (_Float16)(v.z * rn); h.w = (_Float16)(v.w * rn);
  *(v4h*)(pn + (size_t)row * DDIM + lane * 4) = h;
}

// ---------------------------------------------------------------------------
// K1b: class sums S_c[d] = sum_{y2[j]==c} Pn[j][d] and class counts.
// 64 blocks; block handles 128 rows; thread = (half h, column d).
// ---------------------------------------------------------------------------
__global__ __launch_bounds__(256) void scl_classsums(
    const _Float16* __restrict__ pn, const int* __restrict__ y,
    float* __restrict__ S)
{
  const int d  = threadIdx.x & 127;
  const int h  = threadIdx.x >> 7;
  const int r0 = blockIdx.x * 128 + h * 64;

  float a0 = 0.0f, a1 = 0.0f, c0 = 0.0f, c1 = 0.0f;
  #pragma unroll 4
  for (int i = 0; i < 64; ++i) {
    const int row = r0 + i;
    const float v = (float)pn[(size_t)row * DDIM + d];
    const int lab = y[row & (BROWS - 1)];
    a0 += (lab == 0) ? v : 0.0f;
    a1 += (lab == 1) ? v : 0.0f;
    c0 += (lab == 0) ? 1.0f : 0.0f;
    c1 += (lab == 1) ? 1.0f : 0.0f;
  }
  atomicAdd(&S[d],       a0);
  atomicAdd(&S[128 + d], a1);
  if (d == 0) { atomicAdd(&S[256], c0); atomicAdd(&S[257], c1); }
}

// ---------------------------------------------------------------------------
// K2: streaming FULL row sums  denomFull_i = sum_j exp(sim_ij)  (diag incl;
// subtracted analytically in K3). grid = (PPART, NROWS/128): block owns 128
// rows (wave w -> row tile rt = strip*8+w); partition p covers
// jt in {2p+16i, 2p+1+16i : i=0..31}. 32-row (two-tile) B stages through
// LDS, double-buffered, ONE barrier per 2 tiles. WMMA f16, K=128 = 4x32.
// Branchless epilogue: mul+exp+add only.
// ---------------------------------------------------------------------------
__global__ __launch_bounds__(256) void scl_denom(
    const _Float16* __restrict__ pn, float* __restrict__ denom)
{
  __shared__ _Float16 ldsB[2][32 * LDS_STRIDE];   // 2 x 8.5 KiB

  const int tid  = threadIdx.x;
  const int wave = tid >> 5;
  const int lane = tid & 31;
  const int l15  = lane & 15;
  const int lh   = lane >> 4;
  const int p    = blockIdx.x;             // jt partition
  const int rt   = blockIdx.y * 8 + wave;  // this wave's row-tile index

  // A fragments: wave's 16 rows, K=128 in 4 chunks of 32.
  v16h aF[4];
  {
    const _Float16* arow = pn + (size_t)(rt * 16 + l15) * DDIM;
    #pragma unroll
    for (int c = 0; c < 4; ++c) {
      const int kb = c * 32 + lh * 8;
      F16x16 f;
      f.h[0] = *(const v8h*)(arow + kb);
      f.h[1] = *(const v8h*)(arow + kb + 16);
      aF[c] = f.v;
    }
  }

  // cooperative staging of a 32-row (8 KiB) B block: thread loads 32B
  const int srow = tid >> 3;               // 0..31
  const int scol = (tid & 7) * 16;         // halfs
  const _Float16* pf = pn + (size_t)(32 * p + srow) * DDIM + scol;
  v8h st0 = *(const v8h*)(pf);
  v8h st1 = *(const v8h*)(pf + 8);

  float accD[8];
  #pragma unroll
  for (int r = 0; r < 8; ++r) accD[r] = 0.0f;

  #pragma unroll 1
  for (int i = 0; i < NITER; ++i) {
    const int buf = i & 1;
    _Float16* sp = &ldsB[buf][srow * LDS_STRIDE + scol];
    *(v8h*)(sp)     = st0;
    *(v8h*)(sp + 8) = st1;

    // branchless prefetch: uniform step (0 on last iteration -> dead reload)
    const size_t step = (i + 1 < NITER) ? (size_t)(256 * DDIM) : 0;
    pf += step;
    st0 = *(const v8h*)(pf);
    st1 = *(const v8h*)(pf + 8);

    __syncthreads();

    #pragma unroll
    for (int sub = 0; sub < 2; ++sub) {
      const _Float16* bbase =
          &ldsB[buf][(sub * 16 + l15) * LDS_STRIDE + lh * 8];
      v16h bF[4];
      #pragma unroll
      for (int k = 0; k < 4; ++k) {
        F16x16 f;
        f.h[0] = *(const v8h*)(bbase + k * 32);
        f.h[1] = *(const v8h*)(bbase + k * 32 + 16);
        bF[k] = f.v;
      }
      v8f c8 = {};
      #pragma unroll
      for (int k = 0; k < 4; ++k)
        c8 = __builtin_amdgcn_wmma_f32_16x16x32_f16(
            false, aF[k], false, bF[k], (short)0, c8, false, false);
      #pragma unroll
      for (int r = 0; r < 8; ++r)
        accD[r] += __builtin_amdgcn_exp2f(c8[r] * EXP2_SCALE);
    }
  }

  // reduce across the 16 lanes of each half-wave, then one atomic per row
  #pragma unroll
  for (int r = 0; r < 8; ++r) {
    #pragma unroll
    for (int m = 1; m < 16; m <<= 1) accD[r] += __shfl_xor(accD[r], m);
  }
  if (l15 == 0) {
    #pragma unroll
    for (int r = 0; r < 8; ++r)
      atomicAdd(&denom[rt * 16 + r + lh * 8], accD[r]);
  }
}

// ---------------------------------------------------------------------------
// K3: per-row loss term.
//   dP = pn_i . pn_i  (f32)  ->  diag term exp(10*dP), subtract from denomFull
//   s_pos_i = 10*(pn_i . S_{y_i} - dP),  n_pos_i = cnt_{y_i} - 1
//   term = log(denomFull_i - exp(10*dP)) - s_pos_i/n_pos_i
// One wave32 per row; scalar atomicAdd into out.
// ---------------------------------------------------------------------------
__global__ __launch_bounds__(256) void scl_finish(
    const _Float16* __restrict__ pn, const int* __restrict__ y,
    const float* __restrict__ denom, const float* __restrict__ S,
    float* __restrict__ out)
{
  const int wave = threadIdx.x >> 5;
  const int lane = threadIdx.x & 31;
  const int row  = blockIdx.x * 8 + wave;

  const int c = y[row & (BROWS - 1)];
  v4h hv = *(const v4h*)(pn + (size_t)row * DDIM + lane * 4);
  v4f sv = *(const v4f*)(S + c * 128 + lane * 4);

  float dS = 0.0f, dP = 0.0f;
  #pragma unroll
  for (int e = 0; e < 4; ++e) {
    const float f = (float)hv[e];
    dS += f * sv[e];
    dP += f * f;
  }
  #pragma unroll
  for (int m = 1; m < 32; m <<= 1) { dS += __shfl_xor(dS, m); dP += __shfl_xor(dP, m); }

  if (lane == 0) {
    const float diag  = __builtin_amdgcn_exp2f(dP * EXP2_SCALE);
    const float dAdj  = denom[row] - diag;
    const float npos  = S[256 + c] - 1.0f;          // cnt[c] - 1
    const float spos  = 10.0f * (dS - dP);
    const float term  = __builtin_amdgcn_logf(dAdj) * LN2 - spos / npos;
    atomicAdd(out, term);
  }
}

// ---------------------------------------------------------------------------
extern "C" void kernel_launch(void* const* d_in, const int* in_sizes, int n_in,
                              void* d_out, int out_size, void* d_ws, size_t ws_size,
                              hipStream_t stream) {
  const float* zi = (const float*)d_in[0];   // (4096,128) f32
  const float* zj = (const float*)d_in[1];   // (4096,128) f32
  const int*   y  = (const int*)d_in[2];     // (4096,)   i32
  float* out      = (float*)d_out;           // scalar f32 loss

  _Float16* pn  = WS_PN(d_ws);
  float* denom  = WS_DENOM(d_ws);
  float* S      = WS_S(d_ws);                // S[0..255], cnt at S[256..257]

  scl_normalize<<<NROWS / 8, 256, 0, stream>>>(zi, zj, pn, denom, S, out);
  scl_classsums<<<NROWS / 128, 256, 0, stream>>>(pn, y, S);
  dim3 g2(PPART, NROWS / 128);
  scl_denom<<<g2, 256, 0, stream>>>(pn, denom);
  scl_finish<<<NROWS / 8, 256, 0, stream>>>(pn, y, denom, S, out);
}
